// EncoderLayer_83631603188296
// MI455X (gfx1250) — matrix operand order repaired and checked by reading, not compiled, not run on hardware
//
#include <hip/hip_runtime.h>
#include <hip/hip_bf16.h>
#include <cstdint>

typedef __bf16 bf16;
typedef __attribute__((ext_vector_type(16))) __bf16 v16bf;
typedef __attribute__((ext_vector_type(8)))  float  v8f;

#define BB  2
#define SS  2048
#define DD  1024
#define HH  16
#define FFD 4096
#define DHD 64
#define MMR (BB * SS)   // 4096 rows

union PackBF2 { bf16 h[2]; uint32_t u; };

// gfx1250 async global->LDS copy (ASYNCcnt-tracked), see CDNA5 ISA
__device__ __forceinline__ void async_load_b128(const void* gptr, void* lptr) {
    uint32_t lds = (uint32_t)(uintptr_t)lptr;   // low 32 bits of generic ptr = LDS offset
    asm volatile("global_load_async_to_lds_b128 %0, %1, off"
                 :: "v"(lds), "v"(gptr) : "memory");
}
__device__ __forceinline__ void wait_async0() { asm volatile("s_wait_asynccnt 0x0" ::: "memory"); }
__device__ __forceinline__ void wait_async1() { asm volatile("s_wait_asynccnt 0x1" ::: "memory"); }
__device__ __forceinline__ void wait_async2() { asm volatile("s_wait_asynccnt 0x2" ::: "memory"); }

// ---------------- fp32 -> bf16 converter ----------------
__global__ __launch_bounds__(256) void cvt_f32_bf16(const float* __restrict__ in,
                                                    bf16* __restrict__ out, int n) {
    int i = blockIdx.x * blockDim.x + threadIdx.x;
    int stride = gridDim.x * blockDim.x;
    for (; i < n; i += stride) out[i] = (bf16)in[i];
}

// ---------------- generic bf16 WMMA GEMM (double-buffered) ----------------
// C[M,N] = A[M,K] * B[K,N] + bias[N] (+ resid[M,N]) (+ relu)
// block tile 128x128, 8 waves each computing 32x64 (2x4 WMMA 16x16 tiles).
// A tile: async DMA to LDS (ping-pong).  B tile: transposed [n][k] via staged regs.
__global__ __launch_bounds__(256) void gemm_bf16_wmma(
    const bf16* __restrict__ A, const bf16* __restrict__ Bw,
    const float* __restrict__ bias, const float* __restrict__ resid,
    float* __restrict__ Cf, bf16* __restrict__ Cb,
    int M, int N, int K, int relu)
{
    __shared__ __align__(16) bf16 As[2][128][32];
    __shared__ __align__(16) bf16 Bst[2][128][48];  // [n][k], 96B row stride

    const int tid  = threadIdx.x;
    const int lane = tid & 31;
    const int wave = tid >> 5;
    const int wm   = wave >> 1;      // 0..3
    const int wn   = wave & 1;       // 0..1
    const int half = lane >> 4;
    const int l16  = lane & 15;
    const int bm   = blockIdx.x * 128;
    const int bn   = blockIdx.y * 128;

    v8f acc[2][4];
    #pragma unroll
    for (int mi = 0; mi < 2; ++mi)
        #pragma unroll
        for (int ni = 0; ni < 4; ++ni) { v8f z = {}; acc[mi][ni] = z; }

    // loader coordinates (straight-line, no loops)
    const int ar  = tid >> 2, acol = (tid & 3) << 3;          // A rows ar, ar+64
    const int br  = (tid >> 4) << 1, bcol = (tid & 15) << 3;  // B row pair br,br+1
    const bf16* aRow0 = &A[(size_t)(bm + ar) * K + acol];
    const bf16* aRow1 = &A[(size_t)(bm + ar + 64) * K + acol];
    const bf16* bRow  = &Bw[(size_t)br * N + bn + bcol];

    // ---- prologue: stage tile k0 = 0 ----
    async_load_b128(aRow0, &As[0][ar][acol]);
    async_load_b128(aRow1, &As[0][ar + 64][acol]);
    uint4 bu0 = *reinterpret_cast<const uint4*>(bRow);
    uint4 bu1 = *reinterpret_cast<const uint4*>(bRow + N);
    {
        const bf16* p0 = reinterpret_cast<const bf16*>(&bu0);
        const bf16* p1 = reinterpret_cast<const bf16*>(&bu1);
        #pragma unroll
        for (int t = 0; t < 8; ++t) {
            PackBF2 pk; pk.h[0] = p0[t]; pk.h[1] = p1[t];
            *reinterpret_cast<uint32_t*>(&Bst[0][bcol + t][br]) = pk.u;
        }
    }

    for (int k0 = 0; k0 < K; k0 += 32) {
        const int  cur  = (k0 >> 5) & 1;
        const int  nxt  = cur ^ 1;
        const bool more = (k0 + 32) < K;

        if (more) {
            // issue next tile: async A DMA + stage next B rows in registers
            async_load_b128(aRow0 + k0 + 32, &As[nxt][ar][acol]);
            async_load_b128(aRow1 + k0 + 32, &As[nxt][ar + 64][acol]);
            bu0 = *reinterpret_cast<const uint4*>(bRow + (size_t)(k0 + 32) * N);
            bu1 = *reinterpret_cast<const uint4*>(bRow + (size_t)(k0 + 33) * N);
            wait_async2();    // older pair (cur tile) complete; new pair in flight
        } else {
            wait_async0();
        }
        __syncthreads();

        v16bf af[2], bfv[4];
        #pragma unroll
        for (int mi = 0; mi < 2; ++mi) {
            int mr = wm * 32 + mi * 16 + l16;
            #pragma unroll
            for (int e = 0; e < 16; ++e) {
                int kk = e + ((e >> 3) << 3) + (half << 3);   // contiguous runs of 8
                af[mi][e] = As[cur][mr][kk];
            }
        }
        #pragma unroll
        for (int ni = 0; ni < 4; ++ni) {
            int nc = wn * 64 + ni * 16 + l16;
            #pragma unroll
            for (int e = 0; e < 16; ++e)
                bfv[ni][e] = Bst[cur][nc][(half << 3) + e + ((e >> 3) << 3)];   // two runs of 8 (table order)
        }
        #pragma unroll
        for (int mi = 0; mi < 2; ++mi)
            #pragma unroll
            for (int ni = 0; ni < 4; ++ni)
                acc[mi][ni] = __builtin_amdgcn_wmma_f32_16x16x32_bf16(
                    false, af[mi], false, bfv[ni], (short)0, acc[mi][ni], false, false);
        __syncthreads();

        if (more) {   // scatter staged B registers into next buffer
            const bf16* p0 = reinterpret_cast<const bf16*>(&bu0);
            const bf16* p1 = reinterpret_cast<const bf16*>(&bu1);
            #pragma unroll
            for (int t = 0; t < 8; ++t) {
                PackBF2 pk; pk.h[0] = p0[t]; pk.h[1] = p1[t];
                *reinterpret_cast<uint32_t*>(&Bst[nxt][bcol + t][br]) = pk.u;
            }
        }
    }

    #pragma unroll
    for (int mi = 0; mi < 2; ++mi)
        #pragma unroll
        for (int ni = 0; ni < 4; ++ni)
            #pragma unroll
            for (int j = 0; j < 8; ++j) {
                int row = bm + wm * 32 + mi * 16 + j + half * 8;
                int col = bn + wn * 64 + ni * 16 + l16;
                float v = acc[mi][ni][j] + bias[col];
                size_t o = (size_t)row * N + col;
                if (resid) v += resid[o];
                if (relu)  v = v > 0.0f ? v : 0.0f;
                if (Cf) Cf[o] = v;
                if (Cb) Cb[o] = (bf16)v;
            }
}

// ---------------- flash attention (bf16 WMMA, double-buffered) ----------------
// grid: (S/128, B*H). 8 waves/block, each wave owns 16 query rows.
__global__ __launch_bounds__(256) void attn_wmma(
    const bf16* __restrict__ Q, const bf16* __restrict__ Kk,
    const bf16* __restrict__ V, const int* __restrict__ mask,
    bf16* __restrict__ O)
{
    __shared__ __align__(16) bf16 Kt[2][32][64];    // [key][kd] (row-contiguous frags)
    __shared__ __align__(16) bf16 Vtt[2][64][48];   // [n][k] transposed, padded
    __shared__ __align__(16) bf16 Ps[8][16][32];    // per-wave P relayout

    const int tid  = threadIdx.x;
    const int lane = tid & 31, wave = tid >> 5;
    const int half = lane >> 4, l16 = lane & 15;
    const int bh = blockIdx.y;
    const int b = bh >> 4, h = bh & 15;             // H = 16
    const int q0 = blockIdx.x * 128 + wave * 16;
    const size_t headoff = (size_t)h * DHD;

    // preload Q fragments (16x64 -> two 16x32 A-frags)
    v16bf qf[2];
    {
        const bf16* qrow = Q + ((size_t)b * SS + q0 + l16) * DD + headoff;
        #pragma unroll
        for (int ks = 0; ks < 2; ++ks)
            #pragma unroll
            for (int e = 0; e < 16; ++e) {
                int kk = ks * 32 + e + ((e >> 3) << 3) + (half << 3);
                qf[ks][e] = qrow[kk];
            }
    }

    float rm[8], rl[8];
    v8f o[4];
    #pragma unroll
    for (int j = 0; j < 8; ++j) { rm[j] = -1e30f; rl[j] = 0.0f; }
    #pragma unroll
    for (int ni = 0; ni < 4; ++ni) { v8f z = {}; o[ni] = z; }

    const int kr = tid >> 3, kcol = (tid & 7) << 3;     // K: 32 rows x 64 cols
    const int vr = (tid >> 3) << 1;                     // V row pair (tid < 128)
    const bf16* kBase = &Kk[((size_t)b * SS + kr) * DD + headoff + kcol];
    const bf16* vBase = &V[((size_t)b * SS + vr) * DD + headoff + kcol];

    // ---- prologue: stage tile jt = 0 ----
    async_load_b128(kBase, &Kt[0][kr][kcol]);
    uint4 vu0, vu1;
    if (tid < 128) {
        vu0 = *reinterpret_cast<const uint4*>(vBase);
        vu1 = *reinterpret_cast<const uint4*>(vBase + DD);
        const bf16* p0 = reinterpret_cast<const bf16*>(&vu0);
        const bf16* p1 = reinterpret_cast<const bf16*>(&vu1);
        #pragma unroll
        for (int t = 0; t < 8; ++t) {
            PackBF2 pk; pk.h[0] = p0[t]; pk.h[1] = p1[t];
            *reinterpret_cast<uint32_t*>(&Vtt[0][kcol + t][vr]) = pk.u;
        }
    }

    for (int jt = 0; jt < SS / 32; ++jt) {
        const int  cur  = jt & 1;
        const int  nxt  = cur ^ 1;
        const bool more = (jt + 1) < (SS / 32);

        if (more) {
            async_load_b128(kBase + (size_t)(jt + 1) * 32 * DD, &Kt[nxt][kr][kcol]);
            if (tid < 128) {
                const bf16* vp = vBase + (size_t)(jt + 1) * 32 * DD;
                vu0 = *reinterpret_cast<const uint4*>(vp);
                vu1 = *reinterpret_cast<const uint4*>(vp + DD);
            }
            wait_async1();
        } else {
            wait_async0();
        }
        __syncthreads();

        // scores: 16 queries x 32 keys as two 16x16 accumulators
        v8f s[2];
        #pragma unroll
        for (int nj = 0; nj < 2; ++nj) {
            v8f z = {}; s[nj] = z;
            #pragma unroll
            for (int ks = 0; ks < 2; ++ks) {
                v16bf kb;
                #pragma unroll
                for (int e = 0; e < 16; ++e)
                    kb[e] = Kt[cur][nj * 16 + l16][ks * 32 + (half << 3) + e + ((e >> 3) << 3)];
                s[nj] = __builtin_amdgcn_wmma_f32_16x16x32_bf16(
                    false, qf[ks], false, kb, (short)0, s[nj], false, false);
            }
            int kidx = jt * 32 + nj * 16 + l16;
            int mv = mask[b * SS + kidx];
            #pragma unroll
            for (int j = 0; j < 8; ++j) {
                float sv = s[nj][j] * 0.125f;                  // 1/sqrt(64)
                s[nj][j] = (mv == 0) ? -1e9f : sv;
            }
        }

        // online softmax per row (row = 16 lanes of one half-wave)
        #pragma unroll
        for (int j = 0; j < 8; ++j) {
            float tmax = fmaxf(s[0][j], s[1][j]);
            #pragma unroll
            for (int d = 1; d < 16; d <<= 1)
                tmax = fmaxf(tmax, __shfl_xor(tmax, d, 32));
            float mnew = fmaxf(rm[j], tmax);
            float sc = __expf(rm[j] - mnew);
            float p0 = __expf(s[0][j] - mnew);
            float p1 = __expf(s[1][j] - mnew);
            float rs = p0 + p1;
            #pragma unroll
            for (int d = 1; d < 16; d <<= 1)
                rs += __shfl_xor(rs, d, 32);
            rl[j] = rl[j] * sc + rs;
            rm[j] = mnew;
            #pragma unroll
            for (int ni = 0; ni < 4; ++ni) o[ni][j] *= sc;
            Ps[wave][j + half * 8][l16]      = (bf16)p0;       // accum layout -> LDS
            Ps[wave][j + half * 8][16 + l16] = (bf16)p1;
        }

        // P(16x32) x V(32x64) : 4 WMMAs
        v16bf pf;
        #pragma unroll
        for (int e = 0; e < 16; ++e) {
            int kk = e + ((e >> 3) << 3) + (half << 3);
            pf[e] = Ps[wave][l16][kk];
        }
        #pragma unroll
        for (int ni = 0; ni < 4; ++ni) {
            v16bf vbf;
            #pragma unroll
            for (int e = 0; e < 16; ++e)
                vbf[e] = Vtt[cur][ni * 16 + l16][(half << 3) + e + ((e >> 3) << 3)];
            o[ni] = __builtin_amdgcn_wmma_f32_16x16x32_bf16(
                false, pf, false, vbf, (short)0, o[ni], false, false);
        }
        __syncthreads();

        if (more && tid < 128) {   // scatter staged V registers into next buffer
            const bf16* p0 = reinterpret_cast<const bf16*>(&vu0);
            const bf16* p1 = reinterpret_cast<const bf16*>(&vu1);
            #pragma unroll
            for (int t = 0; t < 8; ++t) {
                PackBF2 pk; pk.h[0] = p0[t]; pk.h[1] = p1[t];
                *reinterpret_cast<uint32_t*>(&Vtt[nxt][kcol + t][vr]) = pk.u;
            }
        }
    }

    #pragma unroll
    for (int ni = 0; ni < 4; ++ni)
        #pragma unroll
        for (int j = 0; j < 8; ++j) {
            int row = q0 + j + half * 8;
            int col = ni * 16 + l16;
            O[((size_t)b * SS + row) * DD + headoff + col] = (bf16)(o[ni][j] / rl[j]);
        }
}

// ---------------- layernorm (one block per row of 1024) ----------------
__global__ __launch_bounds__(256) void layernorm_k(
    const float* __restrict__ X, const float* __restrict__ g,
    const float* __restrict__ be, float* __restrict__ Yf,
    bf16* __restrict__ Yb)
{
    const int row = blockIdx.x;
    const int tid = threadIdx.x;
    const float* xr = X + (size_t)row * DD;
    float v[4];
    #pragma unroll
    for (int i = 0; i < 4; ++i) v[i] = xr[tid + i * 256];

    __shared__ float red[8];
    float sum = v[0] + v[1] + v[2] + v[3];
    #pragma unroll
    for (int d = 1; d < 32; d <<= 1) sum += __shfl_xor(sum, d, 32);
    if ((tid & 31) == 0) red[tid >> 5] = sum;
    __syncthreads();
    float mean = 0.0f;
    #pragma unroll
    for (int i = 0; i < 8; ++i) mean += red[i];
    mean *= (1.0f / DD);
    __syncthreads();

    float var = 0.0f;
    #pragma unroll
    for (int i = 0; i < 4; ++i) { float dv = v[i] - mean; var += dv * dv; }
    #pragma unroll
    for (int d = 1; d < 32; d <<= 1) var += __shfl_xor(var, d, 32);
    if ((tid & 31) == 0) red[tid >> 5] = var;
    __syncthreads();
    float vs = 0.0f;
    #pragma unroll
    for (int i = 0; i < 8; ++i) vs += red[i];
    vs *= (1.0f / DD);
    float rstd = rsqrtf(vs + 1e-5f);

    #pragma unroll
    for (int i = 0; i < 4; ++i) {
        int col = tid + i * 256;
        float y = (v[i] - mean) * rstd * g[col] + be[col];
        size_t o = (size_t)row * DD + col;
        if (Yf) Yf[o] = y;
        if (Yb) Yb[o] = (bf16)y;
    }
}

// ---------------- host orchestration ----------------
extern "C" void kernel_launch(void* const* d_in, const int* in_sizes, int n_in,
                              void* d_out, int out_size, void* d_ws, size_t ws_size,
                              hipStream_t stream) {
    const float* x    = (const float*)d_in[0];
    const int*   mask = (const int*)  d_in[1];
    const float* wq   = (const float*)d_in[2];
    const float* bq   = (const float*)d_in[3];
    const float* wk   = (const float*)d_in[4];
    const float* bk   = (const float*)d_in[5];
    const float* wv   = (const float*)d_in[6];
    const float* bv   = (const float*)d_in[7];
    const float* w1   = (const float*)d_in[8];
    const float* b1   = (const float*)d_in[9];
    const float* g1   = (const float*)d_in[10];
    const float* be1  = (const float*)d_in[11];
    const float* fw1  = (const float*)d_in[12];
    const float* fb1  = (const float*)d_in[13];
    const float* fw2  = (const float*)d_in[14];
    const float* fb2  = (const float*)d_in[15];
    const float* g2   = (const float*)d_in[16];
    const float* be2  = (const float*)d_in[17];
    float* out = (float*)d_out;

    char* ws = (char*)d_ws;
    const size_t MB = 1ull << 20;
    bf16*  xb   = (bf16*)(ws + 0);        // 8MB  [4096,1024]
    bf16*  qb   = (bf16*)(ws + 8*MB);     // 8MB
    bf16*  kb   = (bf16*)(ws + 16*MB);    // 8MB
    bf16*  vb   = (bf16*)(ws + 24*MB);    // 8MB
    bf16*  ab   = (bf16*)(ws + 32*MB);    // 8MB  attention out
    bf16*  x2b  = (bf16*)(ws + 40*MB);    // 8MB
    bf16*  wqb  = (bf16*)(ws + 48*MB);    // 2MB
    bf16*  wkb  = (bf16*)(ws + 50*MB);    // 2MB
    bf16*  wvb  = (bf16*)(ws + 52*MB);    // 2MB
    bf16*  w1b  = (bf16*)(ws + 54*MB);    // 2MB
    bf16*  fw1b = (bf16*)(ws + 56*MB);    // 8MB
    bf16*  fw2b = (bf16*)(ws + 64*MB);    // 8MB
    float* y1   = (float*)(ws + 72*MB);   // 16MB (reused for y2)
    float* x2   = (float*)(ws + 88*MB);   // 16MB
    bf16*  hb   = (bf16*)(ws + 8*MB);     // 32MB, reuses dead q/k/v/attn region

    dim3 blk(256);
    cvt_f32_bf16<<<4096, blk, 0, stream>>>(x,   xb,   MMR * DD);
    cvt_f32_bf16<<<2048, blk, 0, stream>>>(wq,  wqb,  DD * DD);
    cvt_f32_bf16<<<2048, blk, 0, stream>>>(wk,  wkb,  DD * DD);
    cvt_f32_bf16<<<2048, blk, 0, stream>>>(wv,  wvb,  DD * DD);
    cvt_f32_bf16<<<2048, blk, 0, stream>>>(w1,  w1b,  DD * DD);
    cvt_f32_bf16<<<4096, blk, 0, stream>>>(fw1, fw1b, DD * FFD);
    cvt_f32_bf16<<<4096, blk, 0, stream>>>(fw2, fw2b, FFD * DD);

    dim3 gD(MMR / 128, DD / 128);    // (32, 8)
    dim3 gF(MMR / 128, FFD / 128);   // (32, 32)

    // QKV projections -> bf16
    gemm_bf16_wmma<<<gD, blk, 0, stream>>>(xb, wqb, bq, nullptr, nullptr, qb, MMR, DD, DD, 0);
    gemm_bf16_wmma<<<gD, blk, 0, stream>>>(xb, wkb, bk, nullptr, nullptr, kb, MMR, DD, DD, 0);
    gemm_bf16_wmma<<<gD, blk, 0, stream>>>(xb, wvb, bv, nullptr, nullptr, vb, MMR, DD, DD, 0);

    // attention
    attn_wmma<<<dim3(SS / 128, BB * HH), blk, 0, stream>>>(qb, kb, vb, mask, ab);

    // out proj + residual, layernorm1
    gemm_bf16_wmma<<<gD, blk, 0, stream>>>(ab, w1b, b1, x, y1, nullptr, MMR, DD, DD, 0);
    layernorm_k<<<MMR, blk, 0, stream>>>(y1, g1, be1, x2, x2b);

    // FFN
    gemm_bf16_wmma<<<gF, blk, 0, stream>>>(x2b, fw1b, fb1, nullptr, nullptr, hb, MMR, FFD, DD, 1);
    gemm_bf16_wmma<<<gD, blk, 0, stream>>>(hb, fw2b, fb2, x2, y1, nullptr, MMR, DD, FFD, 0);
    layernorm_k<<<MMR, blk, 0, stream>>>(y1, g2, be2, out, nullptr);
}
